// DecoderRNN_75539884802147
// MI455X (gfx1250) — compile-verified, matrix operand order checked by reading, not run on hardware
//
#include <hip/hip_runtime.h>

// ---------------- problem constants ----------------
#define Bsz  64
#define NF   196
#define ENCD 512
#define NP   64
#define BENCD 768
#define SS   32
#define VV   30000
#define EE   512
#define DD   512
#define XW   2304   // E + ENC + BENC + D   (x layout: [emb | ctx | pctx | h])
#define GW   2048   // 4*D

typedef __attribute__((ext_vector_type(16))) __bf16 bf16x16;
typedef __attribute__((ext_vector_type(8)))  float  f32x8;
typedef __attribute__((ext_vector_type(4)))  int    i32x4;

union FragBF { uint4 u[2]; bf16x16 v; };

#if __has_builtin(__builtin_amdgcn_global_load_async_to_lds_b128)
#define ASYNC_LDS 1
#else
#define ASYNC_LDS 0
#endif

// ---------------------------------------------------------------
// WMMA bf16 GEMM:  C[M,N] (f32) = A[M,K] @ B[N,K]^T + bias[N]
// grid.x = N/16, grid.y = M/64, block = 128 (4 waves; waves split M,
// share one N-tile). The shared 16xK B tile is staged ONCE into LDS
// via async global->LDS (ASYNCcnt), then each wave's B fragments come
// from ds_load_b128 while A streams from global with prefetch.
// Fragment layout per CDNA5 16-bit A/B spec:
//   row = lane&15, kbase = (lane<16)?0:8, K = {kb..kb+7, kb+16..kb+23}
// ---------------------------------------------------------------
__global__ __launch_bounds__(128)
void wmma_gemm_bf16(const __bf16* __restrict__ A, int lda,
                    const __bf16* __restrict__ Bm, int ldb,
                    float* __restrict__ C, long ldc,
                    const float* __restrict__ bias, int K)
{
    extern __shared__ __bf16 sB[];            // [16][K] row-major
    const int tid  = threadIdx.x;
    const int lane = tid & 31;
    const int wave = tid >> 5;
    const int n0 = blockIdx.x * 16;
    const int m0 = (blockIdx.y * 4 + wave) * 16;

    // ---- stage B tile (16 x K bf16) into LDS, 16B chunk per thread ----
    const int rowChunks = K >> 3;             // 16B chunks per row
    const int nChunks   = rowChunks << 4;     // 2*K, multiple of 128 for our K's
    for (int c = tid; c < nChunks; c += 128) {
        const int brow = c / rowChunks;
        const int bcol = (c - brow * rowChunks) << 3;
        __bf16* src = const_cast<__bf16*>(Bm) + (long)(n0 + brow) * ldb + bcol;
        __bf16* dst = sB + c * 8;             // == brow*K + bcol
#if ASYNC_LDS
        __builtin_amdgcn_global_load_async_to_lds_b128(
            (__attribute__((address_space(1))) i32x4*)src,
            (__attribute__((address_space(3))) i32x4*)dst, 0, 0);
#else
        *(uint4*)dst = *(const uint4*)src;
#endif
    }
#if ASYNC_LDS
#if __has_builtin(__builtin_amdgcn_s_wait_asynccnt)
    __builtin_amdgcn_s_wait_asynccnt(0);
#else
    asm volatile("s_wait_asynccnt 0x0" ::: "memory");
#endif
#endif
    __syncthreads();

    const int r  = lane & 15;
    const int kb = (lane >> 4) << 3;          // 0 or 8
    const __bf16* pa = A  + (long)(m0 + r) * lda + kb;
    const __bf16* pl = sB + (long)r * K + kb;

    f32x8 acc = {0.f,0.f,0.f,0.f,0.f,0.f,0.f,0.f};
    for (int k = 0; k < K; k += 32) {
        FragBF fa, fb;
        fa.u[0] = *(const uint4*)(pa);
        fa.u[1] = *(const uint4*)(pa + 16);
        fb.u[0] = *(const uint4*)(pl);        // ds_load_b128 from staged tile
        fb.u[1] = *(const uint4*)(pl + 16);
        __builtin_prefetch(pa + 64, 0, 3);    // near-cache prefetch, two K slabs ahead
        acc = __builtin_amdgcn_wmma_f32_16x16x32_bf16(
                  false, fa.v, false, fb.v, (short)0, acc, false, false);
        pa += 32; pl += 32;
    }
    const int n  = n0 + r;
    const float bv = bias[n];
    const int mo = m0 + ((lane >> 4) << 3);   // lanes 16..31 hold M=8..15
#pragma unroll
    for (int i = 0; i < 8; ++i)
        C[(long)(mo + i) * ldc + n] = acc[i] + bv;
}

// ---------------- setup helpers ----------------
__global__ void f32_to_bf16_k(const float* __restrict__ in, __bf16* __restrict__ out, long n) {
    long i = (long)blockIdx.x * blockDim.x + threadIdx.x;
    long st = (long)gridDim.x * blockDim.x;
    for (; i < n; i += st) out[i] = (__bf16)in[i];
}

// fuse [lstm_w_ih | lstm_w_hh] -> Wcomb[2048, 2304] (bf16)
__global__ void build_lstm_w_k(const float* __restrict__ wih, const float* __restrict__ whh,
                               __bf16* __restrict__ wc) {
    long i = (long)blockIdx.x * blockDim.x + threadIdx.x;
    long st = (long)gridDim.x * blockDim.x;
    const long total = (long)GW * XW;
    for (; i < total; i += st) {
        int rr = (int)(i / XW), c = (int)(i % XW);
        float v = (c < (EE + ENCD + BENCD)) ? wih[(long)rr * (EE + ENCD + BENCD) + c]
                                            : whh[(long)rr * DD + (c - (EE + ENCD + BENCD))];
        wc[i] = (__bf16)v;
    }
}

__global__ void add_bias_k(const float* a, const float* b, float* o, int n) {
    int i = blockIdx.x * blockDim.x + threadIdx.x;
    if (i < n) o[i] = a[i] + b[i];
}

// mean-pool features & articles -> inp0 bf16 [B, 1280]
__global__ void mean_pool_k(const float* __restrict__ f, const float* __restrict__ a,
                            __bf16* __restrict__ inp0) {
    int b = blockIdx.x;
    for (int d = threadIdx.x; d < ENCD; d += blockDim.x) {
        float s = 0.f;
        for (int n = 0; n < NF; ++n) s += f[((long)b * NF + n) * ENCD + d];
        inp0[(long)b * (ENCD + BENCD) + d] = (__bf16)(s * (1.0f / NF));
    }
    for (int d = threadIdx.x; d < BENCD; d += blockDim.x) {
        float s = 0.f;
        for (int n = 0; n < NP; ++n) s += a[((long)b * NP + n) * BENCD + d];
        inp0[(long)b * (ENCD + BENCD) + ENCD + d] = (__bf16)(s * (1.0f / NP));
    }
}

__global__ void h_to_x_k(const float* __restrict__ h, __bf16* __restrict__ x) {
    int b = blockIdx.x;
    for (int d = threadIdx.x; d < DD; d += blockDim.x)
        x[(long)b * XW + (EE + ENCD + BENCD) + d] = (__bf16)h[(long)b * DD + d];
}

// ---------------- per-step kernels ----------------
__global__ void gather_emb_k(const float* __restrict__ emb, const int* __restrict__ cap,
                             __bf16* __restrict__ x, int t) {
    int b = blockIdx.x;
    int idx = cap[b * SS + t];
    for (int e = threadIdx.x; e < EE; e += blockDim.x)
        x[(long)b * XW + e] = (__bf16)emb[(long)idx * EE + e];
}

// scores[b,n] = tanh(u[b,n,:] + wa[b,waOff:]) . Aw + Ab   (one block per (n,b))
__global__ __launch_bounds__(128)
void attn_score_k(const float* __restrict__ u, const float* __restrict__ wa, int waOff,
                  const float* __restrict__ Aw, const float* __restrict__ Ab,
                  float* __restrict__ scores, int N)
{
    int n = blockIdx.x, b = blockIdx.y;
    const float* up = u + ((long)b * N + n) * 512;
    const float* wp = wa + (long)b * 1024 + waOff;
    float part = 0.f;
    for (int a = threadIdx.x; a < 512; a += 128)
        part += tanhf(up[a] + wp[a]) * Aw[a];
    __shared__ float red[128];
    red[threadIdx.x] = part; __syncthreads();
    for (int s = 64; s > 0; s >>= 1) {
        if (threadIdx.x < s) red[threadIdx.x] += red[threadIdx.x + s];
        __syncthreads();
    }
    if (threadIdx.x == 0) scores[(long)b * N + n] = red[0] + Ab[0];
}

// softmax over N, emit alpha to d_out, ctx (bf16) into x[:, xOff:xOff+Df]
__global__ __launch_bounds__(256)
void softmax_ctx_k(const float* __restrict__ scores, int N,
                   const float* __restrict__ feats, int Df,
                   float* __restrict__ alpha_out, long bStride, int t,
                   __bf16* __restrict__ x, int xOff)
{
    int b = blockIdx.x, tid = threadIdx.x;
    __shared__ float sa[256];
    __shared__ float red[256];
    float v = (tid < N) ? scores[(long)b * N + tid] : -3.0e38f;
    sa[tid] = v; red[tid] = v; __syncthreads();
    for (int s = 128; s > 0; s >>= 1) {
        if (tid < s) red[tid] = fmaxf(red[tid], red[tid + s]);
        __syncthreads();
    }
    float mx = red[0]; __syncthreads();
    float e = (tid < N) ? __expf(sa[tid] - mx) : 0.f;
    red[tid] = e; __syncthreads();
    for (int s = 128; s > 0; s >>= 1) {
        if (tid < s) red[tid] += red[tid + s];
        __syncthreads();
    }
    float al = e / red[0];
    sa[tid] = al;
    if (tid < N) alpha_out[(long)b * bStride + (long)t * N + tid] = al;
    __syncthreads();
    for (int d = tid; d < Df; d += 256) {
        float acc = 0.f;
        for (int n = 0; n < N; ++n)
            acc += sa[n] * feats[((long)b * N + n) * Df + d];
        x[(long)b * XW + xOff + d] = (__bf16)acc;
    }
}

// LSTM pointwise: c (f32, persistent) updated; h written as bf16 into x h-slot
__global__ void lstm_point_k(const float* __restrict__ gates, float* __restrict__ c,
                             __bf16* __restrict__ x)
{
    int b = blockIdx.x;
    for (int d = threadIdx.x; d < DD; d += blockDim.x) {
        float gi = gates[(long)b * GW + d];
        float gf = gates[(long)b * GW + DD + d];
        float gg = gates[(long)b * GW + 2 * DD + d];
        float go = gates[(long)b * GW + 3 * DD + d];
        float si = 1.f / (1.f + __expf(-gi));
        float sf = 1.f / (1.f + __expf(-gf));
        float so = 1.f / (1.f + __expf(-go));
        float cn = sf * c[(long)b * DD + d] + si * tanhf(gg);
        c[(long)b * DD + d] = cn;
        x[(long)b * XW + (EE + ENCD + BENCD) + d] = (__bf16)(so * tanhf(cn));
    }
}

// ---------------------------------------------------------------
extern "C" void kernel_launch(void* const* d_in, const int* in_sizes, int n_in,
                              void* d_out, int out_size, void* d_ws, size_t ws_size,
                              hipStream_t stream) {
    const float* features = (const float*)d_in[0];
    const float* articles = (const float*)d_in[1];
    const int*   captions = (const int*)  d_in[2];
    const float* emb      = (const float*)d_in[3];
    const float* att_W_w  = (const float*)d_in[4];
    const float* att_W_b  = (const float*)d_in[5];
    const float* att_U_w  = (const float*)d_in[6];
    const float* att_U_b  = (const float*)d_in[7];
    const float* att_A_w  = (const float*)d_in[8];
    const float* att_A_b  = (const float*)d_in[9];
    const float* pa_W_w   = (const float*)d_in[10];
    const float* pa_W_b   = (const float*)d_in[11];
    const float* pa_U_w   = (const float*)d_in[12];
    const float* pa_U_b   = (const float*)d_in[13];
    const float* pa_A_w   = (const float*)d_in[14];
    const float* pa_A_b   = (const float*)d_in[15];
    const float* init_h_w = (const float*)d_in[16];
    const float* init_h_b = (const float*)d_in[17];
    const float* init_c_w = (const float*)d_in[18];
    const float* init_c_b = (const float*)d_in[19];
    const float* lstm_w_ih= (const float*)d_in[20];
    const float* lstm_w_hh= (const float*)d_in[21];
    const float* lstm_b_ih= (const float*)d_in[22];
    const float* lstm_b_hh= (const float*)d_in[23];
    const float* fcn_w    = (const float*)d_in[24];
    const float* fcn_b    = (const float*)d_in[25];

    float* outPred  = (float*)d_out;                      // [B,S,V]
    float* outAlpha = outPred  + (long)Bsz * SS * VV;     // [B,S,196]
    float* outPA    = outAlpha + (long)Bsz * SS * NF;     // [B,S,64]

    // ---- workspace carve-up (deterministic) ----
    size_t off = 0;
    auto alloc = [&](size_t bytes) -> void* {
        void* p = (char*)d_ws + off;
        off += (bytes + 255) & ~(size_t)255;
        return p;
    };
    __bf16* featB   = (__bf16*)alloc((size_t)Bsz * NF * ENCD * 2);
    __bf16* artB    = (__bf16*)alloc((size_t)Bsz * NP * BENCD * 2);
    __bf16* attUwB  = (__bf16*)alloc((size_t)512 * ENCD * 2);
    __bf16* paUwB   = (__bf16*)alloc((size_t)512 * BENCD * 2);
    __bf16* attWwB  = (__bf16*)alloc((size_t)512 * DD * 2);
    __bf16* paWwB   = (__bf16*)alloc((size_t)512 * DD * 2);
    __bf16* initHwB = (__bf16*)alloc((size_t)DD * (ENCD + BENCD) * 2);
    __bf16* initCwB = (__bf16*)alloc((size_t)DD * (ENCD + BENCD) * 2);
    __bf16* fcnwB   = (__bf16*)alloc((size_t)VV * DD * 2);
    __bf16* wcombB  = (__bf16*)alloc((size_t)GW * XW * 2);
    float*  biasCmb = (float*) alloc((size_t)GW * 4);
    __bf16* inp0B   = (__bf16*)alloc((size_t)Bsz * (ENCD + BENCD) * 2);
    float*  uFeat   = (float*) alloc((size_t)Bsz * NF * 512 * 4);
    float*  uArt    = (float*) alloc((size_t)Bsz * NP * 512 * 4);
    __bf16* xB      = (__bf16*)alloc((size_t)Bsz * XW * 2);
    float*  wa      = (float*) alloc((size_t)Bsz * 1024 * 4);
    float*  scF     = (float*) alloc((size_t)Bsz * NF * 4);
    float*  scA     = (float*) alloc((size_t)Bsz * NP * 4);
    float*  gatesB  = (float*) alloc((size_t)Bsz * GW * 4);
    float*  cbuf    = (float*) alloc((size_t)Bsz * DD * 4);
    float*  htmp    = (float*) alloc((size_t)Bsz * DD * 4);

    auto gemm = [&](const __bf16* A, int lda, const __bf16* Bm, int ldb,
                    float* C, long ldc, const float* bias, int K, int N, int M) {
        size_t shmem = (size_t)16 * (size_t)K * 2;   // staged B tile
        wmma_gemm_bf16<<<dim3(N / 16, M / 64), dim3(128), shmem, stream>>>(
            A, lda, Bm, ldb, C, ldc, bias, K);
    };

    // ---- one-time setup (per launch) ----
    f32_to_bf16_k<<<1024, 256, 0, stream>>>(features, featB, (long)Bsz * NF * ENCD);
    f32_to_bf16_k<<<1024, 256, 0, stream>>>(articles, artB, (long)Bsz * NP * BENCD);
    f32_to_bf16_k<<<256, 256, 0, stream>>>(att_U_w, attUwB, (long)512 * ENCD);
    f32_to_bf16_k<<<256, 256, 0, stream>>>(pa_U_w,  paUwB,  (long)512 * BENCD);
    f32_to_bf16_k<<<256, 256, 0, stream>>>(att_W_w, attWwB, (long)512 * DD);
    f32_to_bf16_k<<<256, 256, 0, stream>>>(pa_W_w,  paWwB,  (long)512 * DD);
    f32_to_bf16_k<<<256, 256, 0, stream>>>(init_h_w, initHwB, (long)DD * (ENCD + BENCD));
    f32_to_bf16_k<<<256, 256, 0, stream>>>(init_c_w, initCwB, (long)DD * (ENCD + BENCD));
    f32_to_bf16_k<<<2048, 256, 0, stream>>>(fcn_w, fcnwB, (long)VV * DD);
    build_lstm_w_k<<<2048, 256, 0, stream>>>(lstm_w_ih, lstm_w_hh, wcombB);
    add_bias_k<<<8, 256, 0, stream>>>(lstm_b_ih, lstm_b_hh, biasCmb, GW);
    mean_pool_k<<<Bsz, 256, 0, stream>>>(features, articles, inp0B);

    // step-invariant U projections (WMMA)
    gemm(featB, ENCD, attUwB, ENCD, uFeat, 512, att_U_b, ENCD, 512, Bsz * NF);
    gemm(artB, BENCD, paUwB, BENCD, uArt, 512, pa_U_b, BENCD, 512, Bsz * NP);

    // h0 / c0
    gemm(inp0B, ENCD + BENCD, initHwB, ENCD + BENCD, htmp, DD, init_h_b, ENCD + BENCD, DD, Bsz);
    gemm(inp0B, ENCD + BENCD, initCwB, ENCD + BENCD, cbuf, DD, init_c_b, ENCD + BENCD, DD, Bsz);
    h_to_x_k<<<Bsz, 256, 0, stream>>>(htmp, xB);

    // ---- sequential decode loop ----
    const __bf16* hSlot = xB + (EE + ENCD + BENCD);   // h lives in x[:, 1792:2304]
    for (int t = 0; t < SS; ++t) {
        gather_emb_k<<<Bsz, 256, 0, stream>>>(emb, captions, xB, t);

        // w_ah for both attentions (M=64, N=512, K=512)
        gemm(hSlot, XW, attWwB, DD, wa,       1024, att_W_b, DD, 512, Bsz);
        gemm(hSlot, XW, paWwB,  DD, wa + 512, 1024, pa_W_b,  DD, 512, Bsz);

        attn_score_k<<<dim3(NF, Bsz), 128, 0, stream>>>(uFeat, wa, 0,   att_A_w, att_A_b, scF, NF);
        attn_score_k<<<dim3(NP, Bsz), 128, 0, stream>>>(uArt,  wa, 512, pa_A_w,  pa_A_b,  scA, NP);

        softmax_ctx_k<<<Bsz, 256, 0, stream>>>(scF, NF, features, ENCD,
                                               outAlpha, (long)SS * NF, t, xB, EE);
        softmax_ctx_k<<<Bsz, 256, 0, stream>>>(scA, NP, articles, BENCD,
                                               outPA, (long)SS * NP, t, xB, EE + ENCD);

        // gates = [emb|ctx|pctx|h] @ [W_ih|W_hh]^T + (b_ih+b_hh)
        gemm(xB, XW, wcombB, XW, gatesB, GW, biasCmb, XW, GW, Bsz);
        lstm_point_k<<<Bsz, 256, 0, stream>>>(gatesB, cbuf, xB);

        // vocabulary projection straight into d_out[b, t, :]
        gemm(hSlot, XW, fcnwB, DD, outPred + (long)t * VV, (long)SS * VV, fcn_b, DD, VV, Bsz);
    }
    (void)in_sizes; (void)n_in; (void)out_size; (void)ws_size;
}
